// SoftExperts_56118042690100
// MI455X (gfx1250) — matrix operand order repaired and checked by reading, not compiled, not run on hardware
//
#include <hip/hip_runtime.h>

// ---------- types ----------
typedef __bf16 bf16_t;
typedef bf16_t v16bf __attribute__((ext_vector_type(16)));
typedef bf16_t bf16x4 __attribute__((ext_vector_type(4)));
typedef float  v8f   __attribute__((ext_vector_type(8)));
typedef float  f32x4 __attribute__((ext_vector_type(4)));

#define T_TOK 8192
#define D_DIM 1024
#define H_DIM 2048
#define E_NUM 8

// LDS A-tile: 64 rows x 32 K bf16, row stride padded to 40 elems (80 B, 16B-aligned,
// 20-bank step => conflict-free b128 row reads), double buffered.
#define A_STRIDE 40

union FragU { v16bf v; f32x4 q[2]; };

// A fragment (16x32 bf16, MxK) read from an LDS row (current 32-K tile):
// lane L: M = L&15, half = L>>4; K = half*8 + [0,8) and 16 + half*8 + [0,8).
__device__ __forceinline__ v16bf lds_a_frag(const bf16_t* row, int half) {
    FragU u;
    u.q[0] = *reinterpret_cast<const f32x4*>(row + half * 8);
    u.q[1] = *reinterpret_cast<const f32x4*>(row + 16 + half * 8);
    return u.v;
}

// B fragment (32x16 bf16, KxN) from a [N,K] row-major weight row:
// lane L: N = L&15, K = half*8 + [0,8) and 16 + half*8 + [0,8).
__device__ __forceinline__ v16bf load_b_frag(const bf16_t* __restrict__ row, int k, int half) {
    FragU u;
    u.q[0] = *reinterpret_cast<const f32x4*>(row + k + half * 8);
    u.q[1] = *reinterpret_cast<const f32x4*>(row + k + 16 + half * 8);
    return u.v;
}

__device__ __forceinline__ v8f wmma_bf16(v16bf a, v16bf b, v8f c) {
    return __builtin_amdgcn_wmma_f32_16x16x32_bf16(false, a, false, b, (short)0, c, false, false);
}

// Async gather of 16 B from global into LDS (ASYNCcnt-tracked), per-lane addresses.
__device__ __forceinline__ void async_copy_b128(unsigned lds_off, const bf16_t* gsrc) {
    asm volatile("global_load_async_to_lds_b128 %0, %1, off"
                 :: "v"(lds_off), "v"((unsigned long long)gsrc)
                 : "memory");
}
__device__ __forceinline__ void wait_async0() {
    asm volatile("s_wait_asynccnt 0x0" ::: "memory");
}
__device__ __forceinline__ unsigned lds_off_of(const void* p) {
    return (unsigned)(unsigned long long)p;   // LDS offset = low 32 bits of the flat address
}

__device__ __forceinline__ float silu_f(float z) {
    return z / (1.0f + __expf(-z));
}

// ---------- small utility kernels ----------
__global__ void moe_init_kernel(int* __restrict__ counts_and_cursors) {
    if (threadIdx.x < 16) counts_and_cursors[threadIdx.x] = 0;
}

__global__ void moe_zero_f32(float* __restrict__ p, int n4) {
    f32x4 z = {0.f, 0.f, 0.f, 0.f};
    int i = blockIdx.x * blockDim.x + threadIdx.x;
    int stride = gridDim.x * blockDim.x;
    f32x4* p4 = reinterpret_cast<f32x4*>(p);
    for (; i < n4; i += stride) p4[i] = z;
}

__global__ void moe_cvt_bf16(const float* __restrict__ src, bf16_t* __restrict__ dst, int n4) {
    int i = blockIdx.x * blockDim.x + threadIdx.x;
    int stride = gridDim.x * blockDim.x;
    const f32x4* s4 = reinterpret_cast<const f32x4*>(src);
    bf16x4* d4 = reinterpret_cast<bf16x4*>(dst);
    for (; i < n4; i += stride) {
        f32x4 v = s4[i];
        bf16x4 o;
        o[0] = (bf16_t)v[0];
        o[1] = (bf16_t)v[1];
        o[2] = (bf16_t)v[2];
        o[3] = (bf16_t)v[3];
        d4[i] = o;
    }
}

// ---------- gating ----------
__global__ __launch_bounds__(256) void moe_gate_kernel(
    const float* __restrict__ x, const float* __restrict__ gw,
    int* __restrict__ eslot, float* __restrict__ ewt, int* __restrict__ counts)
{
    __shared__ float sh[256 * E_NUM];
    const int t = blockIdx.x;
    const int tid = threadIdx.x;
    const float* xr = x + (size_t)t * D_DIM;

    float acc[E_NUM];
#pragma unroll
    for (int e = 0; e < E_NUM; ++e) acc[e] = 0.f;

#pragma unroll
    for (int j = 0; j < D_DIM / 256; ++j) {
        int d = tid + j * 256;
        float xv = xr[d];
#pragma unroll
        for (int e = 0; e < E_NUM; ++e) acc[e] += xv * gw[e * D_DIM + d];
    }
#pragma unroll
    for (int e = 0; e < E_NUM; ++e) sh[tid * E_NUM + e] = acc[e];
    __syncthreads();

    for (int s = 128; s > 0; s >>= 1) {
        if (tid < s) {
#pragma unroll
            for (int e = 0; e < E_NUM; ++e) sh[tid * E_NUM + e] += sh[(tid + s) * E_NUM + e];
        }
        __syncthreads();
    }

    if (tid == 0) {
        float v0 = -3.4e38f; int i0 = 0;
        for (int e = 0; e < E_NUM; ++e) { float v = sh[e]; if (v > v0) { v0 = v; i0 = e; } }
        float v1 = -3.4e38f; int i1 = 0;
        for (int e = 0; e < E_NUM; ++e) { if (e == i0) continue; float v = sh[e]; if (v > v1) { v1 = v; i1 = e; } }
        float ex = __expf(v1 - v0);
        float inv = 1.f / (1.f + ex);
        eslot[2 * t]     = i0;
        eslot[2 * t + 1] = i1;
        ewt[2 * t]       = inv;
        ewt[2 * t + 1]   = ex * inv;
        atomicAdd(&counts[i0], 1);
        atomicAdd(&counts[i1], 1);
    }
}

// ---------- build per-expert assignment lists ----------
__global__ __launch_bounds__(256) void moe_scatter_kernel(
    const int* __restrict__ eslot, const int* __restrict__ counts,
    int* __restrict__ offs, int* __restrict__ cursors, int* __restrict__ alist)
{
    __shared__ int soffs[E_NUM];
    const int tid = threadIdx.x;
    if (tid == 0) {
        int run = 0;
        for (int e = 0; e < E_NUM; ++e) { soffs[e] = run; offs[e] = run; run += counts[e]; }
    }
    __syncthreads();
    for (int a = tid; a < T_TOK * 2; a += 256) {
        int e = eslot[a];
        int pos = atomicAdd(&cursors[e], 1);
        alist[soffs[e] + pos] = a;   // a = token*2 + slot; outputs keyed by this -> order irrelevant
    }
}

// ---------- pass A: h = (x W1^T + b1) * silu(x W2^T + b2) -> hact (bf16) ----------
// block = 256 thr = 8 waves; block tile 64 rows x 128 H-cols; wave tile 32x32 over both W1,W2.
// A (gathered token rows) staged into LDS via async-to-LDS, double-buffered,
// K-loop manually unrolled by buffer parity so each parity has constant LDS pointers.
__global__ __launch_bounds__(256) void moe_ffn_up(
    const bf16_t* __restrict__ xb,
    const bf16_t* __restrict__ w1b, const bf16_t* __restrict__ w2b,
    const float* __restrict__ b1, const float* __restrict__ b2,
    const int* __restrict__ alist, const int* __restrict__ counts, const int* __restrict__ offs,
    bf16_t* __restrict__ hact)
{
    const int e   = blockIdx.z;
    const int cnt = counts[e];
    const int m0  = blockIdx.x * 64;
    if (cnt == 0 || m0 >= cnt) return;
    const int aoff = offs[e];

    __shared__ int spk[64];
    __shared__ bf16_t sA[2][64 * A_STRIDE];
    const int tid = threadIdx.x;
    if (tid < 64) {
        int ai = m0 + tid;
        if (ai >= cnt) ai = cnt - 1;           // clamp: duplicate rows write identical data (benign)
        spk[tid] = alist[aoff + ai];
    }
    __syncthreads();

    // staging: thread -> (row = tid>>2, 16B chunk = tid&3); one async instr per wave per step
    const int srow = tid >> 2, schunk = tid & 3;
    const bf16_t* gsrc = xb + (size_t)(spk[srow] >> 1) * D_DIM + schunk * 8;
    const unsigned sdst0 = lds_off_of(&sA[0][srow * A_STRIDE + schunk * 8]);
    const unsigned sdst1 = lds_off_of(&sA[1][srow * A_STRIDE + schunk * 8]);

    const int wave = tid >> 5, lane = tid & 31;
    const int half = lane >> 4, r = lane & 15;
    const int wrow = (wave >> 2) * 32;                        // 0 or 32
    const int col0 = blockIdx.y * 128 + (wave & 3) * 32 + r;  // global H column
    const int col1 = col0 + 16;

    const bf16_t* w1r0 = w1b + ((size_t)e * H_DIM + col0) * D_DIM;
    const bf16_t* w1r1 = w1b + ((size_t)e * H_DIM + col1) * D_DIM;
    const bf16_t* w2r0 = w2b + ((size_t)e * H_DIM + col0) * D_DIM;
    const bf16_t* w2r1 = w2b + ((size_t)e * H_DIM + col1) * D_DIM;

    const bf16_t* a0p0 = &sA[0][(wrow + r) * A_STRIDE];
    const bf16_t* a1p0 = &sA[0][(wrow + 16 + r) * A_STRIDE];
    const bf16_t* a0p1 = &sA[1][(wrow + r) * A_STRIDE];
    const bf16_t* a1p1 = &sA[1][(wrow + 16 + r) * A_STRIDE];

    v8f z8 = {0.f,0.f,0.f,0.f,0.f,0.f,0.f,0.f};
    v8f c0a0 = z8, c0a1 = z8, c0b0 = z8, c0b1 = z8;   // rows 0..15 : {W1,W2} x {col0,col1}
    v8f c1a0 = z8, c1a1 = z8, c1b0 = z8, c1b1 = z8;   // rows 16..31

    // 8 WMMAs per K-step in two clusters so only 2 B fragments are live at a time
    auto step = [&](const bf16_t* a0p, const bf16_t* a1p, int k) {
        v16bf A0 = lds_a_frag(a0p, half);
        v16bf A1 = lds_a_frag(a1p, half);
        {
            v16bf B10 = load_b_frag(w1r0, k, half);
            v16bf B20 = load_b_frag(w2r0, k, half);
            c0a0 = wmma_bf16(A0, B10, c0a0);
            c1a0 = wmma_bf16(A1, B10, c1a0);
            c0b0 = wmma_bf16(A0, B20, c0b0);
            c1b0 = wmma_bf16(A1, B20, c1b0);
        }
        {
            v16bf B11 = load_b_frag(w1r1, k, half);
            v16bf B21 = load_b_frag(w2r1, k, half);
            c0a1 = wmma_bf16(A0, B11, c0a1);
            c1a1 = wmma_bf16(A1, B11, c1a1);
            c0b1 = wmma_bf16(A0, B21, c0b1);
            c1b1 = wmma_bf16(A1, B21, c1b1);
        }
    };

    constexpr int STEPS = D_DIM / 32;                  // 32 (even)
    async_copy_b128(sdst0, gsrc);                      // prologue: stage tile 0 -> buf0
    for (int kk = 0; kk < STEPS; kk += 2) {
        wait_async0();
        __syncthreads();                               // buf0 resident; buf1 fully consumed
        async_copy_b128(sdst1, gsrc + (kk + 1) * 32);  // stage next -> buf1
        step(a0p0, a1p0, kk * 32);

        wait_async0();
        __syncthreads();                               // buf1 resident; buf0 fully consumed
        if (kk + 2 < STEPS)
            async_copy_b128(sdst0, gsrc + (kk + 2) * 32);
        step(a0p1, a1p1, (kk + 1) * 32);
    }

    const float b1c0 = b1[e * H_DIM + col0], b1c1 = b1[e * H_DIM + col1];
    const float b2c0 = b2[e * H_DIM + col0], b2c1 = b2[e * H_DIM + col1];
#pragma unroll
    for (int i = 0; i < 8; ++i) {
        // C layout: lane col N = lane&15; row M = half*8 + i per row-fragment
        int mloc0 = wrow + half * 8 + i;
        int mloc1 = mloc0 + 16;
        {
            int pk = spk[mloc0];
            bf16_t* hrow = hact + (size_t)pk * H_DIM;
            hrow[col0] = (bf16_t)((c0a0[i] + b1c0) * silu_f(c0b0[i] + b2c0));
            hrow[col1] = (bf16_t)((c0a1[i] + b1c1) * silu_f(c0b1[i] + b2c1));
        }
        {
            int pk = spk[mloc1];
            bf16_t* hrow = hact + (size_t)pk * H_DIM;
            hrow[col0] = (bf16_t)((c1a0[i] + b1c0) * silu_f(c1b0[i] + b2c0));
            hrow[col1] = (bf16_t)((c1a1[i] + b1c1) * silu_f(c1b1[i] + b2c1));
        }
    }
}

// ---------- pass B: out += gate_wt * (h W3^T + b3), atomic f32 add ----------
// block tile 64 rows x 256 D-cols; wave tile 32x64; LDS-staged A (hact rows), parity-unrolled.
__global__ __launch_bounds__(256) void moe_ffn_down(
    const bf16_t* __restrict__ hact, const bf16_t* __restrict__ w3b,
    const float* __restrict__ b3, const float* __restrict__ ewt,
    const int* __restrict__ alist, const int* __restrict__ counts, const int* __restrict__ offs,
    float* __restrict__ out)
{
    const int e   = blockIdx.z;
    const int cnt = counts[e];
    const int m0  = blockIdx.x * 64;
    if (cnt == 0 || m0 >= cnt) return;
    const int aoff = offs[e];

    __shared__ int spk[64];
    __shared__ bf16_t sA[2][64 * A_STRIDE];
    const int tid = threadIdx.x;
    if (tid < 64) {
        int ai = m0 + tid;
        if (ai >= cnt) ai = cnt - 1;
        spk[tid] = alist[aoff + ai];
    }
    __syncthreads();

    const int srow = tid >> 2, schunk = tid & 3;
    const bf16_t* gsrc = hact + (size_t)spk[srow] * H_DIM + schunk * 8;
    const unsigned sdst0 = lds_off_of(&sA[0][srow * A_STRIDE + schunk * 8]);
    const unsigned sdst1 = lds_off_of(&sA[1][srow * A_STRIDE + schunk * 8]);

    const int wave = tid >> 5, lane = tid & 31;
    const int half = lane >> 4, r = lane & 15;
    const int wrow = (wave >> 2) * 32;
    const int cbase = blockIdx.y * 256 + (wave & 3) * 64 + r;
    const int col[4] = { cbase, cbase + 16, cbase + 32, cbase + 48 };

    const bf16_t* w3r[4];
#pragma unroll
    for (int j = 0; j < 4; ++j) w3r[j] = w3b + ((size_t)e * D_DIM + col[j]) * H_DIM;

    const bf16_t* a0p0 = &sA[0][(wrow + r) * A_STRIDE];
    const bf16_t* a1p0 = &sA[0][(wrow + 16 + r) * A_STRIDE];
    const bf16_t* a0p1 = &sA[1][(wrow + r) * A_STRIDE];
    const bf16_t* a1p1 = &sA[1][(wrow + 16 + r) * A_STRIDE];

    v8f z8 = {0.f,0.f,0.f,0.f,0.f,0.f,0.f,0.f};
    v8f c0[4] = { z8, z8, z8, z8 };   // rows 0..15  x col[0..3]
    v8f c1[4] = { z8, z8, z8, z8 };   // rows 16..31 x col[0..3]

    auto step = [&](const bf16_t* a0p, const bf16_t* a1p, int k) {
        v16bf A0 = lds_a_frag(a0p, half);
        v16bf A1 = lds_a_frag(a1p, half);
#pragma unroll
        for (int j = 0; j < 4; ++j) {
            v16bf B = load_b_frag(w3r[j], k, half);
            c0[j] = wmma_bf16(A0, B, c0[j]);
            c1[j] = wmma_bf16(A1, B, c1[j]);
        }
    };

    constexpr int STEPS = H_DIM / 32;                  // 64 (even)
    async_copy_b128(sdst0, gsrc);
    for (int kk = 0; kk < STEPS; kk += 2) {
        wait_async0();
        __syncthreads();
        async_copy_b128(sdst1, gsrc + (kk + 1) * 32);
        step(a0p0, a1p0, kk * 32);

        wait_async0();
        __syncthreads();
        if (kk + 2 < STEPS)
            async_copy_b128(sdst0, gsrc + (kk + 2) * 32);
        step(a0p1, a1p1, (kk + 1) * 32);
    }

    float bias[4];
#pragma unroll
    for (int j = 0; j < 4; ++j) bias[j] = b3[e * D_DIM + col[j]];

#pragma unroll
    for (int i = 0; i < 8; ++i) {
        int mloc0 = wrow + half * 8 + i;
        int mloc1 = mloc0 + 16;
        if (m0 + mloc0 < cnt) {                 // mask clamped rows: atomics must not double-add
            int pk = spk[mloc0];
            float wt = ewt[pk];
            float* orow = out + (size_t)(pk >> 1) * D_DIM;
#pragma unroll
            for (int j = 0; j < 4; ++j)
                atomicAdd(&orow[col[j]], wt * (c0[j][i] + bias[j]));
        }
        if (m0 + mloc1 < cnt) {
            int pk = spk[mloc1];
            float wt = ewt[pk];
            float* orow = out + (size_t)(pk >> 1) * D_DIM;
#pragma unroll
            for (int j = 0; j < 4; ++j)
                atomicAdd(&orow[col[j]], wt * (c1[j][i] + bias[j]));
        }
    }
}

// ---------- launch ----------
extern "C" void kernel_launch(void* const* d_in, const int* in_sizes, int n_in,
                              void* d_out, int out_size, void* d_ws, size_t ws_size,
                              hipStream_t stream) {
    const float* x  = (const float*)d_in[0];
    const float* gw = (const float*)d_in[1];
    const float* w1 = (const float*)d_in[2];
    const float* b1 = (const float*)d_in[3];
    const float* w2 = (const float*)d_in[4];
    const float* b2 = (const float*)d_in[5];
    const float* w3 = (const float*)d_in[6];
    const float* b3 = (const float*)d_in[7];
    float* out = (float*)d_out;

    char* ws = (char*)d_ws;
    size_t o = 0;
    auto take = [&](size_t bytes) -> char* {
        char* p = ws + o;
        o += (bytes + 255) & ~(size_t)255;
        return p;
    };
    const size_t TD  = (size_t)T_TOK * D_DIM;
    const size_t EHD = (size_t)E_NUM * H_DIM * D_DIM;
    bf16_t* xb    = (bf16_t*)take(TD * 2);
    bf16_t* w1b   = (bf16_t*)take(EHD * 2);
    bf16_t* w2b   = (bf16_t*)take(EHD * 2);
    bf16_t* w3b   = (bf16_t*)take(EHD * 2);
    bf16_t* hact  = (bf16_t*)take((size_t)T_TOK * 2 * H_DIM * 2);
    int*    eslot = (int*)take((size_t)T_TOK * 2 * 4);
    float*  ewt   = (float*)take((size_t)T_TOK * 2 * 4);
    int*    alist = (int*)take((size_t)T_TOK * 2 * 4);
    int*    counts  = (int*)take(16 * 4);   // counts[8] + cursors[8] contiguous
    int*    cursors = counts + 8;
    int*    offs    = (int*)take(8 * 4);

    moe_init_kernel<<<1, 32, 0, stream>>>(counts);
    moe_zero_f32<<<2048, 256, 0, stream>>>(out, (int)(TD / 4));

    moe_cvt_bf16<<<4096, 256, 0, stream>>>(x,  xb,  (int)(TD / 4));
    moe_cvt_bf16<<<4096, 256, 0, stream>>>(w1, w1b, (int)(EHD / 4));
    moe_cvt_bf16<<<4096, 256, 0, stream>>>(w2, w2b, (int)(EHD / 4));
    moe_cvt_bf16<<<4096, 256, 0, stream>>>(w3, w3b, (int)(EHD / 4));

    moe_gate_kernel<<<T_TOK, 256, 0, stream>>>(x, gw, eslot, ewt, counts);
    moe_scatter_kernel<<<1, 256, 0, stream>>>(eslot, counts, offs, cursors, alist);

    dim3 gu(T_TOK / 64, H_DIM / 128, E_NUM);   // worst-case tiles; dead blocks exit on count check
    moe_ffn_up<<<gu, 256, 0, stream>>>(xb, w1b, w2b, b1, b2, alist, counts, offs, hact);

    dim3 gd(T_TOK / 64, D_DIM / 256, E_NUM);
    moe_ffn_down<<<gd, 256, 0, stream>>>(hact, w3b, b3, ewt, alist, counts, offs, out);
}
